// LSTMVAEModel_27358941676034
// MI455X (gfx1250) — matrix operand order repaired and checked by reading, not compiled, not run on hardware
//
#include <hip/hip_runtime.h>
#include <stddef.h>

// ---------------- problem constants ----------------
#define Bsz   64     // batch
#define Csz   64     // channels / decoder hidden
#define Tsz   2048   // time steps
#define Hsz   256    // encoder hidden
#define KCAT  320    // concat K = C+H = H+C
#define KT10  10     // KCAT / 32
#define NTE   64     // encoder gate tiles: 4H/16
#define NTD   16     // decoder gate tiles: 4C/16

typedef __attribute__((ext_vector_type(8)))  _Float16 v8h;
typedef __attribute__((ext_vector_type(16))) _Float16 v16h;
typedef __attribute__((ext_vector_type(8)))  float    v8f;

__device__ __forceinline__ v16h frag_from2(v8h lo, v8h hi) {
  return __builtin_shufflevector(lo, hi, 0,1,2,3,4,5,6,7,8,9,10,11,12,13,14,15);
}

// Fragment block: per (block, lane) 16 contiguous f16 (32B) -> 2x b128 loads.
__device__ __forceinline__ v16h load_block_frag(const _Float16* __restrict__ base,
                                                int blk, int lane) {
  const _Float16* p = base + ((size_t)blk * 32 + lane) * 16;
  v8h lo = *(const v8h*)(p);
  v8h hi = *(const v8h*)(p + 8);
  return frag_from2(lo, hi);
}

// A-matrix (16x32 f16) fragment from a row-major f16 matrix in LDS.
// lanes 0-15: rows M=0..15, K=[0..15]; lanes 16-31: K=[16..31].
__device__ __forceinline__ v16h load_afrag(const _Float16* base, int rowstride,
                                           int r0, int c0, int lane) {
  int row = r0 + (lane & 15);
  int cb  = c0 + ((lane & 16) ? 16 : 0);
  const _Float16* p = base + row * rowstride + cb;
  v8h lo = *(const v8h*)(p);
  v8h hi = *(const v8h*)(p + 8);
  return frag_from2(lo, hi);
}

#define WMMA_F32_F16(A, Bm, Cacc) \
  __builtin_amdgcn_wmma_f32_16x16x32_f16(false, (A), false, (Bm), (short)0, (Cacc), false, false)

__device__ __forceinline__ float sigm(float x) { return 1.0f / (1.0f + __expf(-x)); }

// ---------------- prep kernels (grid parallel) ----------------

// Encoder B = [Wih.T ; Whh.T] (320 x 1024), blocked as (nt*10+kt) fragment blocks.
__global__ void prep_encB(const float* __restrict__ Wih, const float* __restrict__ Whh,
                          _Float16* __restrict__ dst) {
  int blk  = blockIdx.x;            // nt*10 + kt, 640 blocks
  int nt   = blk / KT10, kt = blk % KT10;
  int lane = threadIdx.x;
  _Float16* p = dst + ((size_t)blk * 32 + lane) * 16;
  int N = nt * 16 + (lane & 15);
  int kb = kt * 32 + ((lane & 16) ? 16 : 0);
  #pragma unroll
  for (int h = 0; h < 16; ++h) {
    int K = kb + h;
    float v = (K < Csz) ? Wih[(size_t)N * Csz + K] : Whh[(size_t)N * Hsz + (K - Csz)];
    p[h] = (_Float16)v;
  }
}

// Decoder B = [dec_Wih.T ; dec_Whh.T] (320 x 256), 160 blocks.
__global__ void prep_decB(const float* __restrict__ Wih, const float* __restrict__ Whh,
                          _Float16* __restrict__ dst) {
  int blk  = blockIdx.x;
  int nt   = blk / KT10, kt = blk % KT10;
  int lane = threadIdx.x;
  _Float16* p = dst + ((size_t)blk * 32 + lane) * 16;
  int N = nt * 16 + (lane & 15);
  int kb = kt * 32 + ((lane & 16) ? 16 : 0);
  #pragma unroll
  for (int h = 0; h < 16; ++h) {
    int K = kb + h;
    float v = (K < Hsz) ? Wih[(size_t)N * Hsz + K] : Whh[(size_t)N * Csz + (K - Hsz)];
    p[h] = (_Float16)v;
  }
}

// h2i B = h2i_W.T (64 x 256), 32 blocks (nt*2+kt).
__global__ void prep_h2iB(const float* __restrict__ W, _Float16* __restrict__ dst) {
  int blk  = blockIdx.x;
  int nt   = blk >> 1, kt = blk & 1;
  int lane = threadIdx.x;
  _Float16* p = dst + ((size_t)blk * 32 + lane) * 16;
  int N = nt * 16 + (lane & 15);
  int kb = kt * 32 + ((lane & 16) ? 16 : 0);
  #pragma unroll
  for (int h = 0; h < 16; ++h) p[h] = (_Float16)W[(size_t)N * Csz + kb + h];
}

// x_t A-fragments: for each t, 8 blocks (mt*2+kt), A-matrix lane layout.
__global__ void prep_xfrag(const float* __restrict__ x, _Float16* __restrict__ dst) {
  int t    = blockIdx.x;            // 2048 blocks x 256 threads
  int sub  = threadIdx.x >> 5;      // mt*2 + kt  (0..7)
  int lane = threadIdx.x & 31;
  int mt = sub >> 1, kt = sub & 1;
  size_t blk = (size_t)t * 8 + sub;
  _Float16* p = dst + (blk * 32 + lane) * 16;
  int M = mt * 16 + (lane & 15);
  const float* xrow = x + (size_t)M * (Csz * Tsz) + (size_t)t * Csz + kt * 32;
  int off = (lane & 16) ? 16 : 0;
  #pragma unroll
  for (int h = 0; h < 16; ++h) {
    int K = h + off;
    p[h] = (_Float16)xrow[K];
  }
}

__global__ void prep_bias(const float* __restrict__ b1, const float* __restrict__ b2,
                          float* __restrict__ dst, int n) {
  int i = blockIdx.x * blockDim.x + threadIdx.x;
  if (i < n) dst[i] = b1[i] + b2[i];
}

// ---------------- encoder recurrence (persistent, 1 WG, 32 waves) ----------------
// Wave w: col-group a = w&15, row tiles mtb..mtb+1 (processed sequentially to
// keep acc pressure at 32 VGPRs). Gate tiles nt = a+16g -> elementwise in regs.
// Double-buffered h in LDS: read buf[p], write buf[1-p] -> one barrier/step.
__global__ __launch_bounds__(1024, 1)
void encoder_kernel(const _Float16* __restrict__ Xfrag,
                    const _Float16* __restrict__ WencB,
                    const float* __restrict__ ebias,
                    float* __restrict__ hx_out) {
  __shared__ _Float16 Hbuf[2][Bsz * Hsz];   // 2 x 32KB
  int tid  = threadIdx.x;
  int lane = tid & 31;
  int w    = tid >> 5;
  int a    = w & 15;
  int mtb  = (w >> 4) * 2;
  int l15  = lane & 15;
  int hi8  = (lane & 16) ? 8 : 0;
  int col  = a * 16 + l15;

  for (int i = tid; i < Bsz * Hsz; i += 1024) Hbuf[0][i] = (_Float16)0.0f;

  float bias[4];
  #pragma unroll
  for (int g = 0; g < 4; ++g) bias[g] = ebias[(a + 16 * g) * 16 + l15];

  float cst[2][8];
  #pragma unroll
  for (int mi = 0; mi < 2; ++mi)
    #pragma unroll
    for (int r = 0; r < 8; ++r) cst[mi][r] = 0.0f;

  __syncthreads();

  for (int t = 0; t < Tsz; ++t) {
    const _Float16* Hrd = Hbuf[t & 1];
    _Float16*       Hwr = Hbuf[(t + 1) & 1];

    // prefetch x fragments 4 steps ahead (8KB/step)
    if (t + 4 < Tsz) {
      const char* pf = (const char*)(Xfrag + (size_t)(t + 4) * 8 * 512);
      __builtin_prefetch(pf + tid * 8, 0, 1);
    }

    #pragma unroll
    for (int mi = 0; mi < 2; ++mi) {
      v8f acc[4] = {};
      #pragma unroll 1
      for (int kt = 0; kt < 2; ++kt) {        // x part (pre-blocked global)
        v16h af = load_block_frag(Xfrag, t * 8 + (mtb + mi) * 2 + kt, lane);
        #pragma unroll
        for (int g = 0; g < 4; ++g) {
          v16h bf = load_block_frag(WencB, (a + 16 * g) * KT10 + kt, lane);
          acc[g] = WMMA_F32_F16(af, bf, acc[g]);
        }
      }
      #pragma unroll 1
      for (int kt = 0; kt < 8; ++kt) {        // h part (LDS, read buffer p)
        v16h af = load_afrag(Hrd, Hsz, (mtb + mi) * 16, kt * 32, lane);
        #pragma unroll
        for (int g = 0; g < 4; ++g) {
          v16h bf = load_block_frag(WencB, (a + 16 * g) * KT10 + 2 + kt, lane);
          acc[g] = WMMA_F32_F16(af, bf, acc[g]);
        }
      }
      int rowb = (mtb + mi) * 16 + hi8;
      #pragma unroll
      for (int r = 0; r < 8; ++r) {
        float iv = acc[0][r] + bias[0];
        float fv = acc[1][r] + bias[1];
        float gv = acc[2][r] + bias[2];
        float ov = acc[3][r] + bias[3];
        float cn = sigm(fv) * cst[mi][r] + sigm(iv) * tanhf(gv);
        cst[mi][r] = cn;
        float hn = sigm(ov) * tanhf(cn);
        Hwr[(rowb + r) * Hsz + col] = (_Float16)hn;   // write buffer 1-p
        if (t == Tsz - 1) hx_out[(rowb + r) * Hsz + col] = hn;
      }
    }
    __syncthreads();   // single barrier per step (double-buffered state)
  }
}

// ---------------- variational middle (tiny) ----------------
__global__ void mid1(const float* __restrict__ hx,
                     const float* __restrict__ mW, const float* __restrict__ mb,
                     const float* __restrict__ lW, const float* __restrict__ lb,
                     const float* __restrict__ eps, float* __restrict__ z) {
  int b = blockIdx.x, n = threadIdx.x;              // 64 x 256
  const float* h = hx + (size_t)b * Hsz;
  float m = mb[n], l = lb[n];
  for (int k = 0; k < Hsz; ++k) {
    m += h[k] * mW[(size_t)n * Hsz + k];
    l += h[k] * lW[(size_t)n * Hsz + k];
  }
  z[(size_t)b * Hsz + n] = m + __expf(0.5f * l) * eps[(size_t)b * Hsz + n];
}

__global__ void mid2(const float* __restrict__ z,
                     const float* __restrict__ eW, const float* __restrict__ eb,
                     float* __restrict__ zC) {
  int b = blockIdx.x, cc = threadIdx.x;             // 64 x 64
  float s = eb[cc];
  const float* zr = z + (size_t)b * Hsz;
  for (int k = 0; k < Hsz; ++k) s += zr[k] * eW[(size_t)cc * Hsz + k];
  zC[(size_t)b * Csz + cc] = s;
}

// ---------------- decoder recurrence (persistent, 1 WG, 16 waves) ----------------
// Wave w: mt = w>>2, a = w&3. Gate tiles nt = g*4+a; h2i tiles nt = a*4+jj.
// Double-buffered Acat = [inp(256) | h(64)]: read buf[p], write buf[1-p].
// out[b,c'] = h_dec after step s = 2016-32c', channel 63.
__global__ __launch_bounds__(512, 1)
void decoder_kernel(const _Float16* __restrict__ WdecB,
                    const _Float16* __restrict__ Wh2iB,
                    const float* __restrict__ dbias,
                    const float* __restrict__ h2ib,
                    const float* __restrict__ zC,
                    float* __restrict__ out) {
  __shared__ _Float16 Acat[2][Bsz * KCAT];  // 2 x 40KB
  __shared__ _Float16 Hbuf[Bsz * Csz];      // 8KB (h for h2i GEMM)
  int tid  = threadIdx.x;
  int lane = tid & 31;
  int w    = tid >> 5;
  int a    = w & 3;
  int mt   = w >> 2;
  int l15  = lane & 15;
  int hi8  = (lane & 16) ? 8 : 0;

  for (int i = tid; i < Bsz * KCAT; i += 512) {
    int row = i / KCAT, c = i % KCAT;
    Acat[0][i] = (c < Hsz) ? (_Float16)0.0f
                           : (_Float16)zC[(size_t)row * Csz + (c - Hsz)];
  }
  float gbias[4], ibias[4];
  #pragma unroll
  for (int g = 0; g < 4; ++g)  gbias[g] = dbias[(g * 4 + a) * 16 + l15];
  #pragma unroll
  for (int j = 0; j < 4; ++j)  ibias[j] = h2ib[(a * 4 + j) * 16 + l15];

  float cst[8];
  #pragma unroll
  for (int r = 0; r < 8; ++r) cst[r] = 0.0f;

  __syncthreads();

  int rowb = mt * 16 + hi8;
  int j    = a * 16 + l15;

  for (int s = 0; s < Tsz; ++s) {
    const _Float16* Ard = Acat[s & 1];
    _Float16*       Awr = Acat[(s + 1) & 1];

    // gates = [inp | h] @ WdecB
    v8f acc[4] = {};
    #pragma unroll 1
    for (int kt = 0; kt < KT10; ++kt) {
      v16h af = load_afrag(Ard, KCAT, mt * 16, kt * 32, lane);
      #pragma unroll
      for (int g = 0; g < 4; ++g) {
        v16h bf = load_block_frag(WdecB, (g * 4 + a) * KT10 + kt, lane);
        acc[g] = WMMA_F32_F16(af, bf, acc[g]);
      }
    }

    bool emit = ((s & 31) == 0) && (s <= 2016);
    int  cp   = (2016 - s) >> 5;
    #pragma unroll
    for (int r = 0; r < 8; ++r) {
      float iv = acc[0][r] + gbias[0];
      float fv = acc[1][r] + gbias[1];
      float gv = acc[2][r] + gbias[2];
      float ov = acc[3][r] + gbias[3];
      float cn = sigm(fv) * cst[r] + sigm(iv) * tanhf(gv);
      cst[r] = cn;
      float hn = sigm(ov) * tanhf(cn);
      int row = rowb + r;
      Hbuf[row * Csz + j]            = (_Float16)hn;
      Awr[row * KCAT + Hsz + j]      = (_Float16)hn;   // h region, buffer 1-p
      if (emit && j == 63) out[(size_t)row * Csz + cp] = hn;
    }
    __syncthreads();   // Hbuf fully written before h2i reads it

    // inp = h @ h2i_W.T + h2i_b  -> inp region of buffer 1-p
    v8f pacc[4] = {};
    #pragma unroll 1
    for (int kt = 0; kt < 2; ++kt) {
      v16h af = load_afrag(Hbuf, Csz, mt * 16, kt * 32, lane);
      #pragma unroll
      for (int jj = 0; jj < 4; ++jj) {
        v16h bf = load_block_frag(Wh2iB, (a * 4 + jj) * 2 + kt, lane);
        pacc[jj] = WMMA_F32_F16(af, bf, pacc[jj]);
      }
    }
    #pragma unroll
    for (int jj = 0; jj < 4; ++jj) {
      int coli = (a * 4 + jj) * 16 + l15;
      #pragma unroll
      for (int r = 0; r < 8; ++r)
        Awr[(rowb + r) * KCAT + coli] = (_Float16)(pacc[jj][r] + ibias[jj]);
    }
    __syncthreads();   // buffer 1-p complete before next step
  }
}

// ---------------- launch ----------------
extern "C" void kernel_launch(void* const* d_in, const int* in_sizes, int n_in,
                              void* d_out, int out_size, void* d_ws, size_t ws_size,
                              hipStream_t stream) {
  (void)in_sizes; (void)n_in; (void)out_size; (void)ws_size;
  const float* x        = (const float*)d_in[0];
  const float* eps      = (const float*)d_in[1];
  const float* enc_Wih  = (const float*)d_in[2];
  const float* enc_Whh  = (const float*)d_in[3];
  const float* enc_bih  = (const float*)d_in[4];
  const float* enc_bhh  = (const float*)d_in[5];
  const float* mean_W   = (const float*)d_in[6];
  const float* mean_b   = (const float*)d_in[7];
  const float* lv_W     = (const float*)d_in[8];
  const float* lv_b     = (const float*)d_in[9];
  const float* e2d_W    = (const float*)d_in[10];
  const float* e2d_b    = (const float*)d_in[11];
  const float* dec_Wih  = (const float*)d_in[12];
  const float* dec_Whh  = (const float*)d_in[13];
  const float* dec_bih  = (const float*)d_in[14];
  const float* dec_bhh  = (const float*)d_in[15];
  const float* h2i_W    = (const float*)d_in[16];
  const float* h2i_b    = (const float*)d_in[17];
  float* out = (float*)d_out;

  char* ws = (char*)d_ws;
  _Float16* WencB = (_Float16*)(ws + 0);            //  655360 B
  _Float16* WdecB = (_Float16*)(ws + 655360);       //  163840 B
  _Float16* Wh2iB = (_Float16*)(ws + 819200);       //   32768 B
  float*    ebias = (float*)   (ws + 851968);       //    4096 B
  float*    dbias = (float*)   (ws + 856064);       //    1024 B
  float*    hx    = (float*)   (ws + 857088);       //   65536 B
  float*    z     = (float*)   (ws + 922624);       //   65536 B
  float*    zC    = (float*)   (ws + 988160);       //   16384 B
  _Float16* Xfrag = (_Float16*)(ws + 1004544);      // 16777216 B  (total ~17.0 MB)

  prep_encB<<<NTE * KT10, 32, 0, stream>>>(enc_Wih, enc_Whh, WencB);
  prep_decB<<<NTD * KT10, 32, 0, stream>>>(dec_Wih, dec_Whh, WdecB);
  prep_h2iB<<<NTD * 2, 32, 0, stream>>>(h2i_W, Wh2iB);
  prep_xfrag<<<Tsz, 256, 0, stream>>>(x, Xfrag);
  prep_bias<<<4, 256, 0, stream>>>(enc_bih, enc_bhh, ebias, 4 * Hsz);
  prep_bias<<<1, 256, 0, stream>>>(dec_bih, dec_bhh, dbias, 4 * Csz);

  encoder_kernel<<<1, 1024, 0, stream>>>(Xfrag, WencB, ebias, hx);
  mid1<<<Bsz, Hsz, 0, stream>>>(hx, mean_W, mean_b, lv_W, lv_b, eps, z);
  mid2<<<Bsz, Csz, 0, stream>>>(z, e2d_W, e2d_b, zC);
  decoder_kernel<<<1, 512, 0, stream>>>(WdecB, Wh2iB, dbias, h2i_b, zC, out);
}
